// SelfAttention_14516989461343
// MI455X (gfx1250) — matrix operand order repaired and checked by reading, not compiled, not run on hardware
//
#include <hip/hip_runtime.h>

// ---------------------------------------------------------------------------
// Types for CDNA5 WMMA (gfx1250, wave32)
// ---------------------------------------------------------------------------
typedef __bf16 bf16_t;
typedef __attribute__((ext_vector_type(16))) __bf16 v16bf;
typedef __attribute__((ext_vector_type(8)))  __bf16 v8bf;
typedef __attribute__((ext_vector_type(8)))  float  v8f;
typedef unsigned int u32x4 __attribute__((ext_vector_type(4)));
typedef int          i32x8 __attribute__((ext_vector_type(8)));
typedef int          i32x4 __attribute__((ext_vector_type(4)));

#define N_EMBED 1024
#define N_HEADS 16
#define HEAD_SZ 64
#define SEQ_T   2048
#define BATCH   2
#define MTOT    (BATCH * SEQ_T)   // 4096 rows

#if __has_builtin(__builtin_amdgcn_tensor_load_to_lds) && \
    __has_builtin(__builtin_amdgcn_s_wait_tensorcnt)
#define USE_TDM 1
#else
#define USE_TDM 0
#endif

#if __has_builtin(__builtin_amdgcn_update_dpp)
#define USE_DPP 1
#else
#define USE_DPP 0
#endif

union FragU { v16bf v; v8bf h[2]; };

__device__ __forceinline__ bf16_t f2bf(float f) { return (bf16_t)f; }

// Build a v16bf fragment from two contiguous 16-byte LDS/global chunks.
__device__ __forceinline__ v16bf ld_frag(const bf16_t* p0, const bf16_t* p1) {
  FragU f;
  f.h[0] = *(const v8bf*)p0;
  f.h[1] = *(const v8bf*)p1;
  return f.v;
}

// ---------------------------------------------------------------------------
// Cross-lane reductions over 16-lane rows.
// DPP16 ROW_XMASK (0x160|mask) permutes lane -> lane^mask inside a row of 16;
// clang fuses update_dpp with the consumer into v_max_f32_dpp / v_add_f32_dpp,
// keeping the whole butterfly in the VALU (no ds_bpermute LDS traffic).
// ---------------------------------------------------------------------------
#if USE_DPP
template <int CTRL>
__device__ __forceinline__ float dpp_permf(float x) {
  return __builtin_bit_cast(float,
      __builtin_amdgcn_update_dpp(0, __builtin_bit_cast(int, x),
                                  CTRL, 0xf, 0xf, true));
}
__device__ __forceinline__ float red_max16(float x) {
  x = fmaxf(x, dpp_permf<0x161>(x));   // xor 1
  x = fmaxf(x, dpp_permf<0x162>(x));   // xor 2
  x = fmaxf(x, dpp_permf<0x164>(x));   // xor 4
  x = fmaxf(x, dpp_permf<0x168>(x));   // xor 8
  return x;
}
__device__ __forceinline__ float red_sum16(float x) {
  x += dpp_permf<0x161>(x);
  x += dpp_permf<0x162>(x);
  x += dpp_permf<0x164>(x);
  x += dpp_permf<0x168>(x);
  return x;
}
#else
__device__ __forceinline__ float red_max16(float x) {
  #pragma unroll
  for (int off = 8; off >= 1; off >>= 1) x = fmaxf(x, __shfl_xor(x, off, 16));
  return x;
}
__device__ __forceinline__ float red_sum16(float x) {
  #pragma unroll
  for (int off = 8; off >= 1; off >>= 1) x += __shfl_xor(x, off, 16);
  return x;
}
#endif

#if USE_TDM
// 32-bit LDS byte offset of a __shared__ object (escapes ptr into descriptor).
__device__ __forceinline__ unsigned lds_offset(const void* p) {
  return (unsigned)(unsigned long long)
         (__attribute__((address_space(3))) const char*)p;
}

// TDM: DMA a [rows x tile_w] bf16 tile (row stride 1024 elems in memory) into
// LDS at lds_off with per-row padding (pad_amt+1 DWORDs after each
// 2^(pad_ivl+1) DWORDs) to reproduce the bank-conflict-free LDS pitches.
// D# per CDNA5 ISA 8.3/8.4; amdgpu-toolchain 6-arg builtin form.
__device__ __forceinline__ void tdm_load_tile(const bf16_t* gsrc,
                                              unsigned lds_off, int tile_w,
                                              int rows, int pad_ivl, int pad_amt) {
  unsigned long long ga = (unsigned long long)gsrc;
  u32x4 g0;
  g0[0] = 1u;                                        // count=1, user mode
  g0[1] = lds_off;                                   // lds_addr (bytes)
  g0[2] = (unsigned)(ga & 0xFFFFFFFFu);              // global_addr[31:0]
  g0[3] = (unsigned)((ga >> 32) & 0x01FFFFFFu)       // global_addr[56:32]
        | (2u << 30);                                // type=2 ("image")
  i32x8 g1;
  g1[0] = (1 << 16)                                  // data_size: 2 bytes
        | (1 << 20)                                  // pad_enable
        | (pad_ivl << 22)                            // pad interval (DW, log2-1)
        | (pad_amt << 25);                           // pad amount (DW, minus 1)
  g1[1] = (int)(1024u << 16);                        // tensor_dim0=1024 @bit48
  g1[2] = (int)((unsigned)rows << 16);               // dim0 hi=0 | tensor_dim1 lo
  g1[3] = (int)(((unsigned)rows >> 16)               // tensor_dim1 hi
        | ((unsigned)tile_w << 16));                 // tile_dim0
  g1[4] = rows;                                      // tile_dim1, tile_dim2=0
  g1[5] = 1024;                                      // tensor_dim0_stride lo
  g1[6] = 0;                                         // stride hi | dim1_stride lo
  g1[7] = 0;
  i32x4 z4 = {0, 0, 0, 0};
  i32x8 z8 = {0, 0, 0, 0, 0, 0, 0, 0};
  __builtin_amdgcn_tensor_load_to_lds(g0, g1, z4, z4, z8, 0);
}
#endif

// ---------------------------------------------------------------------------
// Kernel 1: fp32 -> bf16 conversion (x)
// ---------------------------------------------------------------------------
__global__ void cvt_f32_bf16(const float* __restrict__ in,
                             bf16_t* __restrict__ out, int n) {
  int idx = (blockIdx.x * blockDim.x + threadIdx.x) * 4;
  if (idx + 3 < n) {
    float4 f = *(const float4*)(in + idx);
    out[idx + 0] = f2bf(f.x);
    out[idx + 1] = f2bf(f.y);
    out[idx + 2] = f2bf(f.z);
    out[idx + 3] = f2bf(f.w);
  }
}

// ---------------------------------------------------------------------------
// Kernel 2: transpose 1024x1024 fp32 weight -> bf16 [N][K], optional scale.
//   (scale=0.125 folds the attention 1/sqrt(D) into Wq for free)
// ---------------------------------------------------------------------------
__global__ void transpose_w_bf16(const float* __restrict__ in,
                                 bf16_t* __restrict__ out, float scale) {
  __shared__ float tile[32][33];
  int x  = blockIdx.x * 32 + threadIdx.x;   // input col (d)
  int y0 = blockIdx.y * 32;                 // input row (c)
  #pragma unroll
  for (int j = threadIdx.y; j < 32; j += 8)
    tile[j][threadIdx.x] = in[(size_t)(y0 + j) * N_EMBED + x];
  __syncthreads();
  int ox  = blockIdx.y * 32 + threadIdx.x;  // output col (c)
  int oy0 = blockIdx.x * 32;                // output row (d)
  #pragma unroll
  for (int j = threadIdx.y; j < 32; j += 8)
    out[(size_t)(oy0 + j) * N_EMBED + ox] = f2bf(tile[threadIdx.x][j] * scale);
}

// ---------------------------------------------------------------------------
// Kernel 3: WMMA GEMM  C[M,1024] = A[M,1024](bf16) * Bt[1024,1024](bf16)
//   Template F32OUT: fp32 output + bias, else bf16 output. N=K=1024 fixed.
//   Tiles: 128x128, K-step 32, ping-pong LDS; staging via TDM DMA when
//   available (wave 0 issues tensor_load_to_lds, s_wait_tensorcnt, barrier).
// ---------------------------------------------------------------------------
#define GLDT 40   // padded LDS row stride (elements): conflict-free b128 reads

template <bool F32OUT>
__global__ __launch_bounds__(256) void gemm_wmma(
    const bf16_t* __restrict__ A, const bf16_t* __restrict__ Bt,
    bf16_t* __restrict__ Cb, float* __restrict__ Cf,
    const float* __restrict__ bias)
{
  __shared__ bf16_t As[2][128 * GLDT];
  __shared__ bf16_t Bs[2][128 * GLDT];

  const int tid  = threadIdx.x;
  const int lane = tid & 31, w = tid >> 5;
  const int wm = w >> 2, wn = w & 3;
  const int m0 = blockIdx.y * 128, n0 = blockIdx.x * 128;
  const int lm = lane & 15, kh = lane >> 4;

  v8f acc[4][2];
  #pragma unroll
  for (int i = 0; i < 4; i++)
    #pragma unroll
    for (int j = 0; j < 2; j++) acc[i][j] = {};

#if USE_TDM
  const bf16_t* Atile = A  + (size_t)m0 * 1024;   // tile start, k advances below
  const bf16_t* Btile = Bt + (size_t)n0 * 1024;
  const unsigned ldsA[2] = { lds_offset(&As[0][0]), lds_offset(&As[1][0]) };
  const unsigned ldsB[2] = { lds_offset(&Bs[0][0]), lds_offset(&Bs[1][0]) };
  if (w == 0) {                                   // DMA k-tile 0 into buffer 0
    tdm_load_tile(Atile, ldsA[0], 32, 128, 3, 3); // 64B rows + 16B pad -> GLDT
    tdm_load_tile(Btile, ldsB[0], 32, 128, 3, 3);
    __builtin_amdgcn_s_wait_tensorcnt(0);
  }
  __syncthreads();
#else
  // per-thread staging coords: 2 chunks of 8 bf16; chunk i -> row r0+64*i
  const int r0 = tid >> 2, co = (tid & 3) * 8;
  const bf16_t* a0 = A  + (size_t)(m0 + r0) * 1024 + co;
  const bf16_t* a1 = a0 + 64 * 1024;
  const bf16_t* b0 = Bt + (size_t)(n0 + r0) * 1024 + co;
  const bf16_t* b1 = b0 + 64 * 1024;
  const int s0 = r0 * GLDT + co, s1 = s0 + 64 * GLDT;
  {
    v8bf ra0 = *(const v8bf*)a0, ra1 = *(const v8bf*)a1;
    v8bf rb0 = *(const v8bf*)b0, rb1 = *(const v8bf*)b1;
    *(v8bf*)&As[0][s0] = ra0; *(v8bf*)&As[0][s1] = ra1;
    *(v8bf*)&Bs[0][s0] = rb0; *(v8bf*)&Bs[0][s1] = rb1;
  }
  __syncthreads();
#endif

  int buf = 0;
  for (int k0 = 32; k0 <= 1024; k0 += 32) {
    const bool more = (k0 < 1024);

#if USE_TDM
    if (more && w == 0) {                         // DMA next tile -> other buffer
      tdm_load_tile(Atile + k0, ldsA[buf ^ 1], 32, 128, 3, 3);
      tdm_load_tile(Btile + k0, ldsB[buf ^ 1], 32, 128, 3, 3);
    }
#else
    v8bf ra0, ra1, rb0, rb1;
    if (more) {                                   // prefetch next tile to regs
      ra0 = *(const v8bf*)(a0 + k0); ra1 = *(const v8bf*)(a1 + k0);
      rb0 = *(const v8bf*)(b0 + k0); rb1 = *(const v8bf*)(b1 + k0);
    }
#endif

    const bf16_t* Ab = As[buf];
    const bf16_t* Bb = Bs[buf];
    v16bf af[4], bfr[2];
    #pragma unroll
    for (int mt = 0; mt < 4; mt++) {              // A frag: lane lm = row
      const bf16_t* p = &Ab[(wm * 64 + mt * 16 + lm) * GLDT];
      af[mt] = ld_frag(p + kh * 16, p + kh * 16 + 8);
    }
    #pragma unroll
    for (int nt = 0; nt < 2; nt++) {              // B frag: lane lm = col n
      const bf16_t* p = &Bb[(wn * 32 + nt * 16 + lm) * GLDT];
      bfr[nt] = ld_frag(p + kh * 16, p + kh * 16 + 8);
    }
    #pragma unroll
    for (int mt = 0; mt < 4; mt++)
      #pragma unroll
      for (int nt = 0; nt < 2; nt++)
        acc[mt][nt] = __builtin_amdgcn_wmma_f32_16x16x32_bf16(
            false, af[mt], false, bfr[nt], (short)0, acc[mt][nt], false, false);

    if (more) {
#if USE_TDM
      if (w == 0) __builtin_amdgcn_s_wait_tensorcnt(0);
#else
      const int nb = buf ^ 1;                     // nb unused until barrier
      *(v8bf*)&As[nb][s0] = ra0; *(v8bf*)&As[nb][s1] = ra1;
      *(v8bf*)&Bs[nb][s0] = rb0; *(v8bf*)&Bs[nb][s1] = rb1;
#endif
      __syncthreads();
      buf ^= 1;
    }
  }

  // branch-free epilogue; per-row offsets are compile-time immediates
  const int rsel = (lane >> 4) * 8;               // lanes 16-31 = rows 8-15
  #pragma unroll
  for (int mt = 0; mt < 4; mt++) {
    const int grow0 = m0 + wm * 64 + mt * 16 + rsel;
    #pragma unroll
    for (int nt = 0; nt < 2; nt++) {
      const int gcol = n0 + wn * 32 + nt * 16 + lm;
      if constexpr (F32OUT) {
        const float bv = bias[gcol];
        float* cp = Cf + (size_t)grow0 * 1024 + gcol;
        #pragma unroll
        for (int v = 0; v < 8; v++) cp[v * 1024] = acc[mt][nt][v] + bv;
      } else {
        bf16_t* cp = Cb + (size_t)grow0 * 1024 + gcol;
        #pragma unroll
        for (int v = 0; v < 8; v++) cp[v * 1024] = f2bf(acc[mt][nt][v]);
      }
    }
  }
}

// ---------------------------------------------------------------------------
// Kernel 4: flash-style causal attention, all-WMMA.
//   grid = (T/128, H, B). 256 threads = 8 waves; wave owns 16 query rows.
//   1/sqrt(D) pre-folded into Q. Mask applied only on diagonal key blocks.
//   K tile staged by TDM (DMA overlaps the V vector loads); V stored
//   transposed [d][s] to match the B-frag layout for P*V.
// ---------------------------------------------------------------------------
#define KLD 72    // padded stride for K/V tiles
#define PLD 72    // padded stride for per-wave P tile

__global__ __launch_bounds__(256) void attn_wmma(
    const bf16_t* __restrict__ Qm, const bf16_t* __restrict__ Km,
    const bf16_t* __restrict__ Vm, bf16_t* __restrict__ Om)
{
  __shared__ bf16_t Ks[64 * KLD];
  __shared__ bf16_t Vs[64 * KLD];
  __shared__ bf16_t Ps[8 * 16 * PLD];

  const int tid  = threadIdx.x, lane = tid & 31, w = tid >> 5;
  const int qblk = blockIdx.x, h = blockIdx.y, b = blockIdx.z;
  const int lm = lane & 15, kh = lane >> 4;
  const size_t hoff = ((size_t)b * SEQ_T) * N_EMBED + (size_t)h * HEAD_SZ;
  const bf16_t* Qg = Qm + hoff;
  const bf16_t* Kg = Km + hoff;
  const bf16_t* Vg = Vm + hoff;
  bf16_t*       Og = Om + hoff;

  const int qbase = qblk * 128 + w * 16;

  // Q fragments (D=64 -> two 16x32 A-frags), loaded once from global.
  v16bf qf[2];
  #pragma unroll
  for (int ks = 0; ks < 2; ks++) {
    const bf16_t* p = Qg + (size_t)(qbase + lm) * N_EMBED + ks * 32;
    qf[ks] = ld_frag(p + kh * 16, p + kh * 16 + 8);
  }

  v8f o[4];
  #pragma unroll
  for (int ot = 0; ot < 4; ot++) o[ot] = {};
  float m_run[8], l_run[8];
  #pragma unroll
  for (int v = 0; v < 8; v++) { m_run[v] = -1e30f; l_run[v] = 0.f; }

#if USE_TDM
  const unsigned ldsK = lds_offset(&Ks[0]);
#endif
  bf16_t* Pw = &Ps[w * 16 * PLD];
  const int row8 = (lane >> 4) ? 8 : 0;     // this half-wave's row offset
  const int nkb = qblk * 2 + 2;             // causal: key blocks <= diagonal

  for (int kb = 0; kb < nkb; kb++) {
    const int kbase = kb * 64;
    __syncthreads();
#if USE_TDM
    if (w == 0)                             // K tile via TDM: 128B rows + 16B pad
      tdm_load_tile(Kg + (size_t)kbase * 1024, ldsK, 64, 64, 4, 3);
#else
    #pragma unroll
    for (int i = 0; i < 2; i++) {           // K tile: natural [s][d], b128 loads
      int c = tid + i * 256;
      int r = c >> 3, co = (c & 7) * 8;
      *(v8bf*)&Ks[r * KLD + co] = *(const v8bf*)&Kg[(size_t)(kbase + r) * N_EMBED + co];
    }
#endif
    #pragma unroll
    for (int i = 0; i < 2; i++) {           // V tile: b128 load, scatter-transpose
      int c = tid + i * 256;
      int r = c >> 3, co = (c & 7) * 8;
      v8bf vv = *(const v8bf*)&Vg[(size_t)(kbase + r) * N_EMBED + co];
      #pragma unroll
      for (int e = 0; e < 8; e++)
        Vs[(co + e) * KLD + r] = vv[e];
    }
#if USE_TDM
    if (w == 0) __builtin_amdgcn_s_wait_tensorcnt(0);
#endif
    __syncthreads();

    // S = Q * K^T  (k-dim = d, n-dim = s); scale already folded into Q
    v8f sc[4];
    #pragma unroll
    for (int nt = 0; nt < 4; nt++) sc[nt] = {};
    #pragma unroll
    for (int ks = 0; ks < 2; ks++)
      #pragma unroll
      for (int nt = 0; nt < 4; nt++) {
        const bf16_t* p = &Ks[(nt * 16 + lm) * KLD + ks * 32 + kh * 16];
        v16bf bfr = ld_frag(p, p + 8);
        sc[nt] = __builtin_amdgcn_wmma_f32_16x16x32_bf16(
            false, qf[ks], false, bfr, (short)0, sc[nt], false, false);
      }

    // causal mask only on blocks touching the diagonal (wave-uniform branch)
    if (kbase + 63 > qbase) {
      #pragma unroll
      for (int nt = 0; nt < 4; nt++) {
        int kcol = kbase + nt * 16 + lm;
        #pragma unroll
        for (int v = 0; v < 8; v++)
          sc[nt][v] = (kcol <= qbase + row8 + v) ? sc[nt][v] : -1e30f;
      }
    }

    // row max (VALU DPP butterfly over the 16-lane row)
    float rmax[8];
    #pragma unroll
    for (int v = 0; v < 8; v++)
      rmax[v] = red_max16(
          fmaxf(fmaxf(sc[0][v], sc[1][v]), fmaxf(sc[2][v], sc[3][v])));

    // online softmax state update
    float rsum[8];
    #pragma unroll
    for (int v = 0; v < 8; v++) {
      float nm = fmaxf(m_run[v], rmax[v]);
      float alpha = __expf(m_run[v] - nm);
      m_run[v] = nm;
      l_run[v] *= alpha;
      #pragma unroll
      for (int ot = 0; ot < 4; ot++) o[ot][v] *= alpha;
      rsum[v] = 0.f;
    }

    // P = exp(S - m), C-layout -> LDS (per-wave), partial row sums
    #pragma unroll
    for (int nt = 0; nt < 4; nt++)
      #pragma unroll
      for (int v = 0; v < 8; v++) {
        float p = __expf(sc[nt][v] - m_run[v]);
        rsum[v] += p;
        Pw[(v + row8) * PLD + nt * 16 + lm] = f2bf(p);
      }
    #pragma unroll
    for (int v = 0; v < 8; v++) l_run[v] += red_sum16(rsum[v]);

    // O += P * V   (k-dim = s, n-dim = d); P reloaded as A-frags from LDS
    #pragma unroll
    for (int ks = 0; ks < 2; ks++) {
      const bf16_t* pb = &Pw[lm * PLD + ks * 32];
      v16bf pf = ld_frag(pb + kh * 16, pb + kh * 16 + 8);
      #pragma unroll
      for (int ot = 0; ot < 4; ot++) {
        const bf16_t* vp = &Vs[(ot * 16 + lm) * KLD + ks * 32 + kh * 16];
        v16bf vf = ld_frag(vp, vp + 8);
        o[ot] = __builtin_amdgcn_wmma_f32_16x16x32_bf16(
            false, pf, false, vf, (short)0, o[ot], false, false);
      }
    }
  }

  // epilogue: O /= l, write bf16 to [b*T+t][h*64+d]
  #pragma unroll
  for (int v = 0; v < 8; v++) {
    float inv = 1.0f / l_run[v];
    int grow = qbase + row8 + v;
    #pragma unroll
    for (int ot = 0; ot < 4; ot++)
      Og[(size_t)grow * N_EMBED + ot * 16 + lm] = f2bf(o[ot][v] * inv);
  }
}

// ---------------------------------------------------------------------------
// Host-side orchestration
// ---------------------------------------------------------------------------
extern "C" void kernel_launch(void* const* d_in, const int* in_sizes, int n_in,
                              void* d_out, int out_size, void* d_ws, size_t ws_size,
                              hipStream_t stream) {
  const float* x  = (const float*)d_in[0];
  const float* Wk = (const float*)d_in[1];
  const float* Wq = (const float*)d_in[2];
  const float* Wv = (const float*)d_in[3];
  const float* Wp = (const float*)d_in[4];
  const float* bp = (const float*)d_in[5];

  // workspace layout (bytes); total 48 MB
  char* ws = (char*)d_ws;
  const size_t XW  = (size_t)MTOT * N_EMBED * sizeof(bf16_t);      // 8 MB
  const size_t WW  = (size_t)N_EMBED * N_EMBED * sizeof(bf16_t);   // 2 MB
  bf16_t* x_bf = (bf16_t*)(ws);
  bf16_t* WqT  = (bf16_t*)(ws + XW);
  bf16_t* WkT  = (bf16_t*)(ws + XW + WW);
  bf16_t* WvT  = (bf16_t*)(ws + XW + 2 * WW);
  bf16_t* WpT  = (bf16_t*)(ws + XW + 3 * WW);
  bf16_t* Qb   = (bf16_t*)(ws + XW + 4 * WW);
  bf16_t* Kb   = (bf16_t*)(ws + 2 * XW + 4 * WW);
  bf16_t* Vb   = (bf16_t*)(ws + 3 * XW + 4 * WW);
  bf16_t* Ab   = (bf16_t*)(ws + 4 * XW + 4 * WW);

  // 1) convert x to bf16
  {
    int n = MTOT * N_EMBED;
    cvt_f32_bf16<<<(n / 4 + 255) / 256, 256, 0, stream>>>(x, x_bf, n);
  }
  // 2) transpose+convert weights (1/sqrt(D) folded into Wq)
  {
    dim3 g(32, 32), blk(32, 8);
    transpose_w_bf16<<<g, blk, 0, stream>>>(Wq, WqT, 0.125f);
    transpose_w_bf16<<<g, blk, 0, stream>>>(Wk, WkT, 1.0f);
    transpose_w_bf16<<<g, blk, 0, stream>>>(Wv, WvT, 1.0f);
    transpose_w_bf16<<<g, blk, 0, stream>>>(Wp, WpT, 1.0f);
  }
  // 3) Q/K/V projections (bf16 out)
  {
    dim3 g(N_EMBED / 128, MTOT / 128);   // (8, 32)
    gemm_wmma<false><<<g, 256, 0, stream>>>(x_bf, WqT, Qb, nullptr, nullptr);
    gemm_wmma<false><<<g, 256, 0, stream>>>(x_bf, WkT, Kb, nullptr, nullptr);
    gemm_wmma<false><<<g, 256, 0, stream>>>(x_bf, WvT, Vb, nullptr, nullptr);
  }
  // 4) causal flash attention
  {
    dim3 g(SEQ_T / 128, N_HEADS, BATCH); // (16, 16, 2)
    attn_wmma<<<g, 256, 0, stream>>>(Qb, Kb, Vb, Ab);
  }
  // 5) output projection, fp32 + bias
  {
    dim3 g(N_EMBED / 128, MTOT / 128);
    gemm_wmma<true><<<g, 256, 0, stream>>>(Ab, WpT, nullptr, (float*)d_out, bp);
  }
}
